// G2C_8727373545529
// MI455X (gfx1250) — matrix operand order repaired and checked by reading, not compiled, not run on hardware
//
#include <hip/hip_runtime.h>
#include <hip/hip_bf16.h>
#include <math.h>

typedef __attribute__((ext_vector_type(16))) _Float16 v16h;
typedef __attribute__((ext_vector_type(8)))  float    v8f;
typedef unsigned int u32x4 __attribute__((ext_vector_type(4)));
typedef int          i32x8 __attribute__((ext_vector_type(8)));
typedef int          i32x4 __attribute__((ext_vector_type(4)));

#define BATCH 1024
#define NPTS  21
#define NN    441            // 21*21 rows per batch
#define HDIM  256
#define W1T_STRIDE 264       // halves per row: 256 + 8 pad (528 B) -> conflict-free b128 reads

// ---------------------------------------------------------------------------
// Kernel 0: W1 [256,256] fp32 row-major  ->  W1^T fp16, padded rows, in d_ws
// ---------------------------------------------------------------------------
__global__ void prep_w1t(const float* __restrict__ W1, _Float16* __restrict__ w1t) {
    int n = blockIdx.x;
    int k = threadIdx.x;
    w1t[n * W1T_STRIDE + k] = (_Float16)W1[k * HDIM + n];
}

// ---------------------------------------------------------------------------
// Kernel 1: fused  relu(A@W1+b1) -> emb (batch sum), e2=(.)@W2+b2,
//           symmetrize + softplus -> D_init, W.   One workgroup per batch.
//   128 threads = 4 waves; wave w owns row-blocks [7w, 7w+6]:
//   3 dual-block iterations (shared B fragments, 2 indep. WMMA chains) + 1 single.
// ---------------------------------------------------------------------------
__launch_bounds__(128)
__global__ void fused_gemm(const float*    __restrict__ A,    // [B*441, 256]
                           const _Float16* __restrict__ w1t,  // [256, 264] f16
                           const float*    __restrict__ b1,
                           const float*    __restrict__ W2,   // [256, 2]
                           const float*    __restrict__ b2,
                           float* __restrict__ out_D,
                           float* __restrict__ out_W,
                           float* __restrict__ out_emb) {
    extern __shared__ char smem[];
    _Float16* w1t_lds = (_Float16*)smem;                         // 135168 B (offset 0!)
    float*    e2_lds  = (float*)(smem + 135168);                 // 3584 B
    float*    emb_lds = (float*)(smem + 135168 + 3584);          // 1024 B
    float*    w2_lds  = (float*)(smem + 135168 + 3584 + 1024);   // 2048 B
    float*    b1_lds  = (float*)(smem + 135168 + 3584 + 3072);   // 1024 B
    float*    b2_lds  = (float*)(smem + 135168 + 3584 + 4096);   // 8 B

    const int tid  = threadIdx.x;
    const int b    = blockIdx.x;
    const int wave = tid >> 5;
    const int lane = tid & 31;
    const int lo   = lane & 15;
    const int hi   = lane >> 4;

    // ---- stage W1^T into LDS: one Tensor Data Mover descriptor (wave 0) ----
#if __has_builtin(__builtin_amdgcn_tensor_load_to_lds) && __has_builtin(__builtin_amdgcn_s_wait_tensorcnt)
    if (wave == 0) {
        unsigned long long ga = (unsigned long long)(uintptr_t)w1t;
        u32x4 g0;
        g0[0] = 1u;                                   // count=1, user descriptor
        g0[1] = 0u;                                   // lds_addr = 0 (w1t_lds at base)
        g0[2] = (unsigned)(ga & 0xFFFFFFFFu);         // global_addr[31:0]
        g0[3] = (unsigned)((ga >> 32) & 0x1FFFFFFu) | (2u << 30);  // addr[56:32] | type=2
        i32x8 g1;
        g1[0] = (1 << 16);          // workgroup_mask=0, data_size=1 (2 bytes)
        g1[1] = (W1T_STRIDE << 16); // tensor_dim0[15:0] << 16
        g1[2] = (HDIM << 16);       // tensor_dim0 hi = 0 | tensor_dim1[15:0] << 16
        g1[3] = (W1T_STRIDE << 16); // tensor_dim1 hi = 0 | tile_dim0 << 16
        g1[4] = HDIM;               // tile_dim1 | tile_dim2 = 0
        g1[5] = W1T_STRIDE;         // tensor_dim0_stride lo32
        g1[6] = 0;                  // stride hi | tensor_dim1_stride (unused, 2D tile)
        g1[7] = 0;
        i32x4 z4 = {0, 0, 0, 0};                  // groups 2/3: null (<=2D tensor)
        i32x8 z8 = {0, 0, 0, 0, 0, 0, 0, 0};      // trailing group (clang-23 6-arg form)
        __builtin_amdgcn_tensor_load_to_lds(g0, g1, z4, z4, z8, 0);
        __builtin_amdgcn_s_wait_tensorcnt(0);
    }
#else
    {
        const uint4* src = (const uint4*)w1t;
        uint4*       dst = (uint4*)w1t_lds;
        for (int i = tid; i < (HDIM * W1T_STRIDE) / 8; i += 128) dst[i] = src[i];
    }
#endif
    for (int i = tid; i < HDIM; i += 128) {
        emb_lds[i]      = 0.0f;
        b1_lds[i]       = b1[i];
        w2_lds[2*i]     = W2[2*i];
        w2_lds[2*i + 1] = W2[2*i + 1];
    }
    if (tid < 2) b2_lds[tid] = b2[tid];
    __syncthreads();

    // A-fragment loader: WMMA 16x32 f16 A layout.
    // lane<16: K = c*32 + {0..7, 16..23}; lane>=16: K = c*32 + {8..15, 24..31}
    auto loadA = [&](v16h* af, int blk) {
        int rib = blk * 16 + lo; if (rib > NN - 1) rib = NN - 1;
        const float* arow = A + ((size_t)b * NN + rib) * HDIM;
        #pragma unroll
        for (int c = 0; c < 8; ++c) {
            int k0 = c * 32 + hi * 8;
            float4 p0 = *(const float4*)(arow + k0);
            float4 p1 = *(const float4*)(arow + k0 + 4);
            float4 q0 = *(const float4*)(arow + k0 + 16);
            float4 q1 = *(const float4*)(arow + k0 + 20);
            v16h v;
            v[0]=(_Float16)p0.x; v[1]=(_Float16)p0.y; v[2]=(_Float16)p0.z; v[3]=(_Float16)p0.w;
            v[4]=(_Float16)p1.x; v[5]=(_Float16)p1.y; v[6]=(_Float16)p1.z; v[7]=(_Float16)p1.w;
            v[8]=(_Float16)q0.x; v[9]=(_Float16)q0.y; v[10]=(_Float16)q0.z; v[11]=(_Float16)q0.w;
            v[12]=(_Float16)q1.x; v[13]=(_Float16)q1.y; v[14]=(_Float16)q1.z; v[15]=(_Float16)q1.w;
            af[c] = v;
        }
    };
    // B fragment from LDS: lane<16 holds K={0..7,16..23}, lane>=16 holds K={8..15,24..31} of chunk
    auto loadB = [&](v16h* bf, int ct) {
        typedef __attribute__((ext_vector_type(8))) _Float16 v8h_;
        const int ncol = ct * 16 + lo;
        #pragma unroll
        for (int c8 = 0; c8 < 8; ++c8) {
            const _Float16* bp = &w1t_lds[ncol * W1T_STRIDE + c8 * 32 + hi * 8];
            v8h_ lo8 = *(const v8h_*)bp;
            v8h_ hi8 = *(const v8h_*)(bp + 16);
            bf[c8] = __builtin_shufflevector(lo8, hi8, 0,1,2,3,4,5,6,7,8,9,10,11,12,13,14,15);
        }
    };
    auto reduce_store_e2 = [&](float* aa, float* ab, int blk) {
        #pragma unroll
        for (int i = 0; i < 8; ++i) {
            float t0 = aa[i], t1 = ab[i];
            t0 += __shfl_xor(t0, 1, 32);  t1 += __shfl_xor(t1, 1, 32);
            t0 += __shfl_xor(t0, 2, 32);  t1 += __shfl_xor(t1, 2, 32);
            t0 += __shfl_xor(t0, 4, 32);  t1 += __shfl_xor(t1, 4, 32);
            t0 += __shfl_xor(t0, 8, 32);  t1 += __shfl_xor(t1, 8, 32);
            if (lo == 0) {
                const int r = blk * 16 + i + 8 * hi;   // rows > 440 written, never read
                e2_lds[r * 2 + 0] = t0 + b2_lds[0];
                e2_lds[r * 2 + 1] = t1 + b2_lds[1];
            }
        }
    };

    for (int p = 0; p < 4; ++p) {
        const int  blk0 = 7 * wave + 2 * p;     // p==3 -> single block 7w+6
        const bool dual = (p < 3);
        const int  blk1 = blk0 + 1;

        v16h a0[8], a1[8];
        loadA(a0, blk0);
        if (dual) loadA(a1, blk1);

        float m0v[8];
        #pragma unroll
        for (int i = 0; i < 8; ++i)
            m0v[i] = (blk0 * 16 + i + 8 * hi < NN) ? 1.0f : 0.0f;   // emb row mask

        float e0a[8], e0b[8], e1a[8], e1b[8];
        #pragma unroll
        for (int i = 0; i < 8; ++i) { e0a[i]=0.f; e0b[i]=0.f; e1a[i]=0.f; e1b[i]=0.f; }

        v16h bf[8], bn[8];
        loadB(bf, 0);
        #pragma unroll 2
        for (int ct = 0; ct < 16; ++ct) {
            if (ct < 15) loadB(bn, ct + 1);     // prefetch next tile's B fragments
            v8f c0 = {}, c1 = {};
            if (dual) {
                #pragma unroll
                for (int c8 = 0; c8 < 8; ++c8) {
                    c0 = __builtin_amdgcn_wmma_f32_16x16x32_f16(false, a0[c8], false, bf[c8], (short)0, c0, false, false);
                    c1 = __builtin_amdgcn_wmma_f32_16x16x32_f16(false, a1[c8], false, bf[c8], (short)0, c1, false, false);
                }
            } else {
                #pragma unroll
                for (int c8 = 0; c8 < 8; ++c8)
                    c0 = __builtin_amdgcn_wmma_f32_16x16x32_f16(false, a0[c8], false, bf[c8], (short)0, c0, false, false);
            }
            const int   ncol = ct * 16 + lo;
            const float bias = b1_lds[ncol];
            const float w2a  = w2_lds[2 * ncol];
            const float w2b  = w2_lds[2 * ncol + 1];
            float s = 0.0f;
            if (dual) {
                #pragma unroll
                for (int i = 0; i < 8; ++i) {
                    const float d0 = fmaxf(c0[i] + bias, 0.0f);
                    const float d1 = fmaxf(c1[i] + bias, 0.0f);
                    s = fmaf(d0, m0v[i], s);  s += d1;
                    e0a[i] = fmaf(d0, w2a, e0a[i]);  e0b[i] = fmaf(d0, w2b, e0b[i]);
                    e1a[i] = fmaf(d1, w2a, e1a[i]);  e1b[i] = fmaf(d1, w2b, e1b[i]);
                }
            } else {
                #pragma unroll
                for (int i = 0; i < 8; ++i) {
                    const float d0 = fmaxf(c0[i] + bias, 0.0f);
                    s = fmaf(d0, m0v[i], s);
                    e0a[i] = fmaf(d0, w2a, e0a[i]);  e0b[i] = fmaf(d0, w2b, e0b[i]);
                }
            }
            atomicAdd(&emb_lds[ncol], s);       // ds_add_f32
            #pragma unroll
            for (int c8 = 0; c8 < 8; ++c8) bf[c8] = bn[c8];
        }

        reduce_store_e2(e0a, e0b, blk0);
        if (dual) reduce_store_e2(e1a, e1b, blk1);
    }
    __syncthreads();

    // ---- epilogue: symmetrize + softplus -> D_init, W ; flush emb ----
    for (int idx = tid; idx < NN; idx += 128) {
        const int i = idx / NPTS, j = idx % NPTS;
        const float s0 = e2_lds[(i * NPTS + j) * 2 + 0] + e2_lds[(j * NPTS + i) * 2 + 0];
        const float s1 = e2_lds[(i * NPTS + j) * 2 + 1] + e2_lds[(j * NPTS + i) * 2 + 1];
        const float sp0 = (s0 > 20.0f) ? s0 : log1pf(expf(s0));
        const float sp1 = (s1 > 20.0f) ? s1 : log1pf(expf(s1));
        out_D[(size_t)b * NN + idx] = (i == j) ? 0.0f : sp0;
        out_W[(size_t)b * NN + idx] = sp1;
    }
    for (int i = tid; i < HDIM; i += 128) out_emb[(size_t)b * HDIM + i] = emb_lds[i];
}

// ---------------------------------------------------------------------------
// Kernel 2: MDS reconstruction. One wave per molecule (lane = node),
//           8 molecules per 256-thread block. Latency-bound serial chain.
// ---------------------------------------------------------------------------
__launch_bounds__(256)
__global__ void recon(const float* __restrict__ Dall, const float* __restrict__ Wall,
                      const float* __restrict__ u_noise, const float* __restrict__ x_noise,
                      float* __restrict__ Xout) {
    __shared__ float Dl[8][NN];
    __shared__ float Wl[8][NN];
    __shared__ float Al[8][NN];
    __shared__ float uarr[8][32];
    __shared__ float rsarr[8][32];

    const int wave = threadIdx.x >> 5;
    const int lane = threadIdx.x & 31;
    const int b = blockIdx.x * 8 + wave;

    const float* Db = Dall + (size_t)b * NN;
    const float* Wb = Wall + (size_t)b * NN;
    for (int idx = lane; idx < NN; idx += 32) { Dl[wave][idx] = Db[idx]; Wl[wave][idx] = Wb[idx]; }
    __builtin_amdgcn_wave_barrier();

    // ---- classical MDS double-centering (D symmetric: colsum == rowsum) ----
    float rsum = 0.0f;
    if (lane < NPTS) {
        for (int j = 0; j < NPTS; ++j) rsum += Dl[wave][lane * NPTS + j];
    }
    rsarr[wave][lane] = (lane < NPTS) ? rsum : 0.0f;
    float tot = (lane < NPTS) ? rsum : 0.0f;
    tot += __shfl_xor(tot, 1, 32);  tot += __shfl_xor(tot, 2, 32);
    tot += __shfl_xor(tot, 4, 32);  tot += __shfl_xor(tot, 8, 32);
    tot += __shfl_xor(tot, 16, 32);
    __builtin_amdgcn_wave_barrier();
    if (lane < NPTS) {
        const float rm = rsum * (1.0f / NPTS);
        const float gm = tot * (1.0f / (NPTS * NPTS));
        for (int j = 0; j < NPTS; ++j) {
            Al[wave][lane * NPTS + j] =
                -0.5f * (Dl[wave][lane * NPTS + j] - rsarr[wave][j] * (1.0f / NPTS) - rm + gm);
        }
    }
    __builtin_amdgcn_wave_barrier();

    // ---- deflated power iteration, k=3, 10 steps each ----
    float us[3];
    #pragma unroll
    for (int r = 0; r < 3; ++r) {
        float u = (lane < NPTS) ? u_noise[(size_t)b * 3 * NPTS + r * NPTS + lane] : 0.0f;
        for (int step = 0; step < 10; ++step) {
            float nn = (lane < NPTS) ? u * u : 0.0f;
            nn += __shfl_xor(nn, 1, 32);  nn += __shfl_xor(nn, 2, 32);
            nn += __shfl_xor(nn, 4, 32);  nn += __shfl_xor(nn, 8, 32);
            nn += __shfl_xor(nn, 16, 32);
            u = u / fmaxf(sqrtf(nn), 1e-3f);
            uarr[wave][lane] = (lane < NPTS) ? u : 0.0f;
            __builtin_amdgcn_wave_barrier();
            float s = 0.0f;
            if (lane < NPTS)
                for (int j = 0; j < NPTS; ++j) s += Al[wave][lane * NPTS + j] * uarr[wave][j];
            u = s;
            __builtin_amdgcn_wave_barrier();
        }
        float eig = (lane < NPTS) ? u * u : 0.0f;
        eig += __shfl_xor(eig, 1, 32);  eig += __shfl_xor(eig, 2, 32);
        eig += __shfl_xor(eig, 4, 32);  eig += __shfl_xor(eig, 8, 32);
        eig += __shfl_xor(eig, 16, 32);
        u = u / powf(eig + 0.01f, 0.25f);
        us[r] = u;
        uarr[wave][lane] = (lane < NPTS) ? u : 0.0f;
        __builtin_amdgcn_wave_barrier();
        if (lane < NPTS)
            for (int j = 0; j < NPTS; ++j) Al[wave][lane * NPTS + j] -= u * uarr[wave][j];
        __builtin_amdgcn_wave_barrier();
    }

    // ---- X0 = stack(us) + x_noise ----
    float x = 0.f, y = 0.f, z = 0.f;
    if (lane < NPTS) {
        const float* xn = x_noise + (size_t)b * NPTS * 3 + lane * 3;
        x = us[0] + xn[0];  y = us[1] + xn[1];  z = us[2] + xn[2];
    }

    float drow[NPTS], wrow[NPTS];
    #pragma unroll
    for (int j = 0; j < NPTS; ++j) {
        drow[j] = (lane < NPTS) ? Dl[wave][lane * NPTS + j] : 0.0f;
        wrow[j] = (lane < NPTS) ? Wl[wave][lane * NPTS + j] : 0.0f;
    }

    // ---- 100 clipped gradient-descent steps ----
    for (int t = 0; t < 100; ++t) {
        float gx = 0.f, gy = 0.f, gz = 0.f;
        #pragma unroll
        for (int j = 0; j < NPTS; ++j) {
            const float xj = __shfl(x, j, 32);
            const float yj = __shfl(y, j, 32);
            const float zj = __shfl(z, j, 32);
            const float dx = x - xj, dy = y - yj, dz = z - zj;
            const float d2 = dx * dx + dy * dy + dz * dz + 0.01f;
            const float tt = wrow[j] * (drow[j] - d2);   // == W[j][i]*(D[j][i]-d2) by symmetry
            gx += tt * dx;  gy += tt * dy;  gz += tt * dz;
        }
        const float dXx = 0.8f * gx, dXy = 0.8f * gy, dXz = 0.8f * gz;
        const float speed = sqrtf(dXx * dXx + dXy * dXy + dXz * dXz + 1e-3f);
        const float alpha = 0.1f + 4.9f * ((100.0f - (float)t) * 0.01f);
        const float scale = alpha * tanhf(speed / alpha) / speed;
        x += dXx * scale;  y += dXy * scale;  z += dXz * scale;
    }

    if (lane < NPTS) {
        float* o = Xout + (size_t)b * NPTS * 3 + lane * 3;
        o[0] = x; o[1] = y; o[2] = z;
    }
}

// ---------------------------------------------------------------------------
extern "C" void kernel_launch(void* const* d_in, const int* in_sizes, int n_in,
                              void* d_out, int out_size, void* d_ws, size_t ws_size,
                              hipStream_t stream) {
    const float* gnn  = (const float*)d_in[0];
    const float* W1   = (const float*)d_in[1];
    const float* b1   = (const float*)d_in[2];
    const float* W2   = (const float*)d_in[3];
    const float* b2   = (const float*)d_in[4];
    const float* un   = (const float*)d_in[5];
    const float* xn   = (const float*)d_in[6];
    (void)in_sizes; (void)n_in; (void)out_size; (void)ws_size;

    float* out   = (float*)d_out;
    float* out_D = out;
    float* out_W = out + (size_t)BATCH * NN;
    float* out_E = out + (size_t)2 * BATCH * NN;
    float* out_X = out_E + (size_t)BATCH * HDIM;

    _Float16* w1t = (_Float16*)d_ws;                  // 256*264 f16 = 135168 B

    prep_w1t<<<HDIM, HDIM, 0, stream>>>(W1, w1t);

    const size_t smem_bytes = 135168 + 3584 + 1024 + 2048 + 1024 + 8;  // 142856 B
    fused_gemm<<<BATCH, 128, smem_bytes, stream>>>(gnn, w1t, b1, W2, b2, out_D, out_W, out_E);

    recon<<<BATCH / 8, 256, 0, stream>>>(out_D, out_W, un, xn, out_X);
}
